// MTTNet_win2swin_27599459844909
// MI455X (gfx1250) — compile-verified, matrix operand order checked by reading, not run on hardware
//
#include <hip/hip_runtime.h>
#include <hip/hip_bf16.h>
#include <math.h>

// ---------------- static problem config (matches reference) ----------------
#define S3D     16
#define H3D     96
#define W3D     96
#define CDIM    96
#define NH      3
#define HD      32
#define NTOK    64           // tokens per 4x4x4 window
#define NWIN    2304         // (16/4)*(96/4)*(96/4)
#define LTOK    147456       // S*H*W
#define SHIFT   2
#define SCALE   0.17677669529663687f   // 32^-0.5

typedef __attribute__((ext_vector_type(16))) _Float16 v16h;
typedef __attribute__((ext_vector_type(8)))  float    v8f;

// K-element index inside a v16h WMMA fragment for lane-group g (lane>>4),
// per CDNA5 ISA 7.12.2 "16-bit A-Matrix 16x32" layout. B mirrors with
// lane = N column index.
__device__ __forceinline__ int kof(int i, int g) {
  return (i < 8) ? (g * 8 + i) : (8 + g * 8 + i);
}

__device__ __forceinline__ v8f wmma_f16(v16h a, v16h b, v8f c) {
  return __builtin_amdgcn_wmma_f32_16x16x32_f16(false, a, false, b, (short)0, c,
                                                false, false);
}

// ---- CDNA5 async global->LDS copy (ASYNCcnt-tracked, bypasses VGPRs) ----
// 32-bit LDS byte address of a __shared__ object (generic -> as(3) cast).
// Host pass parses __device__ bodies too, so keep the as(3) cast device-only
// and go through size_t (as(3) pointers are 64-bit on host, 32-bit on device).
__device__ __forceinline__ unsigned int lds_addr_of(const void* p) {
#if defined(__HIP_DEVICE_COMPILE__)
#pragma clang diagnostic push
#pragma clang diagnostic ignored "-Wold-style-cast"
  return (unsigned int)(size_t)(__attribute__((address_space(3))) const void*)p;
#pragma clang diagnostic pop
#else
  return (unsigned int)(size_t)p;
#endif
}

// One 16-byte per-lane transfer: LDS[lds_byte_addr] = MEM[gsrc .. gsrc+15]
__device__ __forceinline__ void async_g2l_b128(unsigned int lds_byte_addr,
                                               const void* gsrc) {
  asm volatile("global_load_async_to_lds_b128 %0, %1, off"
               :: "v"(lds_byte_addr), "v"(gsrc)
               : "memory");
}

__device__ __forceinline__ void wait_async0() {
  asm volatile("s_wait_asynccnt 0x0" ::: "memory");
}

// ---------------- weight fp32 -> fp16 conversion ----------------
__global__ __launch_bounds__(256) void cvt_f16_kernel(const float* __restrict__ in,
                                                      _Float16* __restrict__ out,
                                                      int n) {
  int i = blockIdx.x * 256 + threadIdx.x;
  if (i < n) out[i] = (_Float16)in[i];
}

// ---------------- LN1 + cyclic shift + window partition -> xw (f16) ----------
// One wave per destination (window, token). Gathers the shifted source token,
// LayerNorms it (3 channels/lane, shfl reductions) and stores f16 row.
__global__ __launch_bounds__(256) void ln1_shift_part_kernel(
    const float* __restrict__ x, const float* __restrict__ gam,
    const float* __restrict__ bet, _Float16* __restrict__ xw) {
  int wave = threadIdx.x >> 5, lane = threadIdx.x & 31;
  int d = blockIdx.x * 8 + wave;                 // 0 .. NWIN*64-1
  int widx = d >> 6, tidx = d & 63;
  int si = widx / 576, rem = widx % 576, hi = rem / 24, wi = rem % 24;
  int ts = tidx >> 4, th = (tidx >> 2) & 3, tw = tidx & 3;
  int ss = ((si << 2) + ts + SHIFT) & (S3D - 1);
  int sh = ((hi << 2) + th + SHIFT) % H3D;
  int sw = ((wi << 2) + tw + SHIFT) % W3D;
  size_t src = ((size_t)ss * H3D + sh) * W3D + sw;
  const float* xr = x + src * CDIM;
  float v0 = xr[lane], v1 = xr[lane + 32], v2 = xr[lane + 64];
  float s = v0 + v1 + v2, s2 = v0 * v0 + v1 * v1 + v2 * v2;
#pragma unroll
  for (int m = 16; m; m >>= 1) {
    s  += __shfl_xor(s,  m, 32);
    s2 += __shfl_xor(s2, m, 32);
  }
  float mean = s * (1.0f / CDIM);
  float inv  = rsqrtf(s2 * (1.0f / CDIM) - mean * mean + 1e-5f);
  _Float16* o = xw + (size_t)d * CDIM;
  o[lane]      = (_Float16)((v0 - mean) * inv * gam[lane]      + bet[lane]);
  o[lane + 32] = (_Float16)((v1 - mean) * inv * gam[lane + 32] + bet[lane + 32]);
  o[lane + 64] = (_Float16)((v2 - mean) * inv * gam[lane + 64] + bet[lane + 64]);
}

// ---------------- fused per-window attention ----------------
// QKV GEMM -> per-head (QK^T + bias + mask -> softmax -> PV) -> proj ->
// reverse-shift scatter + residual. All intermediates LDS-resident.
__global__ __launch_bounds__(256) void swin_attn_kernel(
    const _Float16* __restrict__ xw,
    const _Float16* __restrict__ qkv_w16, const float* __restrict__ qkv_b,
    const _Float16* __restrict__ proj_w16, const float* __restrict__ proj_b,
    const float* __restrict__ rpb, const int* __restrict__ relidx,
    const float* __restrict__ maskm, const float* __restrict__ x,
    float* __restrict__ x1) {
  // LDS: 64x296 f16 QKV (37888B) | region A: f16 64x104 tile / f32 64x68
  // scores (17408B, phase-aliased) | 64x64 f16 probs (8192B)  => 63488B
  __shared__ _Float16 sQKV[64 * 296];
  __shared__ float    sRegA[64 * 68];
  __shared__ _Float16 sP[64 * 64];
  _Float16* sX = (_Float16*)sRegA;   // stride 104 halfs (208B, 16B-aligned rows)
  float*    sS = sRegA;              // stride 68 floats

  const int win  = blockIdx.x;
  const int wave = threadIdx.x >> 5, lane = threadIdx.x & 31;
  const int g = lane >> 4, l15 = lane & 15;

  // window coordinates for the output scatter
  const int si = win / 576, remw = win % 576, hi = remw / 24, wi = remw % 24;

  // ---- async-stage input tile [64,96] f16 into LDS (768 x B128) ----
  {
    const unsigned int sx_base = lds_addr_of(sX);
    const _Float16* gbase = xw + (size_t)win * (NTOK * CDIM);
#pragma unroll
    for (int t = 0; t < 3; ++t) {
      int k = threadIdx.x + 256 * t;          // 0..767
      int r = k / 12, c = k - r * 12;         // 12 x 16B per row
      async_g2l_b128(sx_base + (unsigned)(r * 104 + c * 8) * 2,
                     gbase + r * CDIM + c * 8);
    }
  }
  wait_async0();
  __syncthreads();

  // ---- QKV: [64,96] x [96,288] ; 72 tiles, 9 per wave ----
  for (int t = 0; t < 9; ++t) {
    int tt = wave + 8 * t;
    int mt = tt & 3, nt = tt >> 2;     // nt 0..17
    int j = nt * 16 + l15;             // output col 0..287 (= 3C layout)
    v8f c;
    float bias = qkv_b[j];
#pragma unroll
    for (int r = 0; r < 8; ++r) c[r] = bias;
#pragma unroll
    for (int kb = 0; kb < CDIM; kb += 32) {
      v16h a, bf;
      const _Float16* wr = qkv_w16 + (size_t)j * CDIM + kb;
      __builtin_prefetch(wr + CDIM, 0, 0);
#pragma unroll
      for (int i = 0; i < 16; ++i) a[i] = sX[(mt * 16 + l15) * 104 + kb + kof(i, g)];
#pragma unroll
      for (int i = 0; i < 16; ++i) bf[i] = wr[kof(i, g)];
      c = wmma_f16(a, bf, c);
    }
    float scale = (j < CDIM) ? SCALE : 1.0f;   // q gets pre-scaled
#pragma unroll
    for (int r = 0; r < 8; ++r) {
      int m = mt * 16 + r + (g << 3);
      sQKV[m * 296 + j] = (_Float16)(c[r] * scale);
    }
  }
  __syncthreads();

  // ---- per-head attention ----
  for (int h = 0; h < NH; ++h) {
    // scores = qk^T + bias + mask : 16 tiles, 2 per wave, K=32 (one WMMA)
    for (int t = 0; t < 2; ++t) {
      int tt = wave + 8 * t;
      int mt = tt & 3, nt = tt >> 2;
      v8f c;
#pragma unroll
      for (int r = 0; r < 8; ++r) {
        int m = mt * 16 + r + (g << 3), n = nt * 16 + l15;
        c[r] = rpb[relidx[m * 64 + n] * NH + h] +
               maskm[(size_t)win * 4096 + m * 64 + n];
      }
      v16h a, bf;
#pragma unroll
      for (int i = 0; i < 16; ++i)
        a[i] = sQKV[(mt * 16 + l15) * 296 + h * HD + kof(i, g)];
#pragma unroll
      for (int i = 0; i < 16; ++i)
        bf[i] = sQKV[(nt * 16 + l15) * 296 + CDIM + h * HD + kof(i, g)];
      c = wmma_f16(a, bf, c);
#pragma unroll
      for (int r = 0; r < 8; ++r) {
        int m = mt * 16 + r + (g << 3), n = nt * 16 + l15;
        sS[m * 68 + n] = c[r];
      }
    }
    __syncthreads();

    // softmax: 8 rows per wave, 2 cols per lane
    for (int rr = 0; rr < 8; ++rr) {
      int row = wave * 8 + rr;
      float v0 = sS[row * 68 + lane], v1 = sS[row * 68 + 32 + lane];
      float mx = fmaxf(v0, v1);
#pragma unroll
      for (int m = 16; m; m >>= 1) mx = fmaxf(mx, __shfl_xor(mx, m, 32));
      float e0 = __expf(v0 - mx), e1 = __expf(v1 - mx);
      float sm = e0 + e1;
#pragma unroll
      for (int m = 16; m; m >>= 1) sm += __shfl_xor(sm, m, 32);
      float inv = 1.0f / sm;
      sP[row * 64 + lane]      = (_Float16)(e0 * inv);
      sP[row * 64 + 32 + lane] = (_Float16)(e1 * inv);
    }
    __syncthreads();

    // O_h = P @ V_h : 8 tiles, 1 per wave, K=64; overwrite Q_h slice
    {
      int mt = wave & 3, nt = wave >> 2;   // nt 0..1
      v8f c = {};
#pragma unroll
      for (int kb = 0; kb < NTOK; kb += 32) {
        v16h a, bf;
#pragma unroll
        for (int i = 0; i < 16; ++i)
          a[i] = sP[(mt * 16 + l15) * 64 + kb + kof(i, g)];
#pragma unroll
        for (int i = 0; i < 16; ++i)
          bf[i] = sQKV[(kb + kof(i, g)) * 296 + 2 * CDIM + h * HD + nt * 16 + l15];
        c = wmma_f16(a, bf, c);
      }
#pragma unroll
      for (int r = 0; r < 8; ++r) {
        int m = mt * 16 + r + (g << 3), n = nt * 16 + l15;
        sQKV[m * 296 + h * HD + n] = (_Float16)c[r];
      }
    }
    __syncthreads();
  }

  // ---- proj: [64,96] x [96,96] + bias, reverse-shift scatter + residual ----
  for (int t = 0; t < 3; ++t) {
    int tt = wave + 8 * t;
    int mt = tt & 3, nt = tt >> 2;     // nt 0..5
    int n = nt * 16 + l15;
    v8f c;
    float bias = proj_b[n];
#pragma unroll
    for (int r = 0; r < 8; ++r) c[r] = bias;
#pragma unroll
    for (int kb = 0; kb < CDIM; kb += 32) {
      v16h a, bf;
      const _Float16* wr = proj_w16 + (size_t)n * CDIM + kb;
      __builtin_prefetch(wr + CDIM, 0, 0);
#pragma unroll
      for (int i = 0; i < 16; ++i) a[i] = sQKV[(mt * 16 + l15) * 296 + kb + kof(i, g)];
#pragma unroll
      for (int i = 0; i < 16; ++i) bf[i] = wr[kof(i, g)];
      c = wmma_f16(a, bf, c);
    }
#pragma unroll
    for (int r = 0; r < 8; ++r) {
      int m = mt * 16 + r + (g << 3);               // token within window
      int ts = m >> 4, th = (m >> 2) & 3, tw = m & 3;
      int ss = ((si << 2) + ts + SHIFT) & (S3D - 1);
      int sh = ((hi << 2) + th + SHIFT) % H3D;
      int sw = ((wi << 2) + tw + SHIFT) % W3D;
      size_t off = (((size_t)ss * H3D + sh) * W3D + sw) * CDIM + n;
      x1[off] = x[off] + c[r];                      // residual 1
    }
  }
}

// ---------------- LN2 (identity mapping) -> f16 ----------------
__global__ __launch_bounds__(256) void ln2_kernel(const float* __restrict__ x1,
                                                  const float* __restrict__ gam,
                                                  const float* __restrict__ bet,
                                                  _Float16* __restrict__ o) {
  int wave = threadIdx.x >> 5, lane = threadIdx.x & 31;
  size_t d = (size_t)blockIdx.x * 8 + wave;
  const float* xr = x1 + d * CDIM;
  float v0 = xr[lane], v1 = xr[lane + 32], v2 = xr[lane + 64];
  float s = v0 + v1 + v2, s2 = v0 * v0 + v1 * v1 + v2 * v2;
#pragma unroll
  for (int m = 16; m; m >>= 1) {
    s  += __shfl_xor(s,  m, 32);
    s2 += __shfl_xor(s2, m, 32);
  }
  float mean = s * (1.0f / CDIM);
  float inv  = rsqrtf(s2 * (1.0f / CDIM) - mean * mean + 1e-5f);
  _Float16* op = o + d * CDIM;
  op[lane]      = (_Float16)((v0 - mean) * inv * gam[lane]      + bet[lane]);
  op[lane + 32] = (_Float16)((v1 - mean) * inv * gam[lane + 32] + bet[lane + 32]);
  op[lane + 64] = (_Float16)((v2 - mean) * inv * gam[lane + 64] + bet[lane + 64]);
}

// ---------------- fc1 (96->384) + exact GELU -> h1 (f16) ----------------
__global__ __launch_bounds__(256) void fc1_gelu_kernel(
    const _Float16* __restrict__ a16, const _Float16* __restrict__ w16,
    const float* __restrict__ bias, _Float16* __restrict__ h1) {
  __shared__ _Float16 sA[64 * 104];     // 16B-aligned rows for async B128
  int rowbase = blockIdx.x * 64;
  {
    const unsigned int sa_base = lds_addr_of(sA);
    const _Float16* gbase = a16 + (size_t)rowbase * CDIM;
#pragma unroll
    for (int t = 0; t < 3; ++t) {
      int k = threadIdx.x + 256 * t;          // 0..767
      int r = k / 12, c = k - r * 12;
      async_g2l_b128(sa_base + (unsigned)(r * 104 + c * 8) * 2,
                     gbase + r * CDIM + c * 8);
    }
  }
  wait_async0();
  __syncthreads();
  int wave = threadIdx.x >> 5, lane = threadIdx.x & 31;
  int g = lane >> 4, l15 = lane & 15;
  for (int t = 0; t < 12; ++t) {
    int tt = wave + 8 * t;              // 96 tiles: 4 M x 24 N
    int mt = tt & 3, nt = tt >> 2;
    int j = nt * 16 + l15;
    v8f c;
    float bv = bias[j];
#pragma unroll
    for (int r = 0; r < 8; ++r) c[r] = bv;
#pragma unroll
    for (int kb = 0; kb < CDIM; kb += 32) {
      v16h a, bf;
      const _Float16* wr = w16 + (size_t)j * CDIM + kb;
      __builtin_prefetch(wr + CDIM, 0, 0);
#pragma unroll
      for (int i = 0; i < 16; ++i) a[i] = sA[(mt * 16 + l15) * 104 + kb + kof(i, g)];
#pragma unroll
      for (int i = 0; i < 16; ++i) bf[i] = wr[kof(i, g)];
      c = wmma_f16(a, bf, c);
    }
#pragma unroll
    for (int r = 0; r < 8; ++r) {
      int m = mt * 16 + r + (g << 3);
      float v = c[r];
      v = 0.5f * v * (1.0f + erff(v * 0.70710678118654752f));   // exact GELU
      h1[(size_t)(rowbase + m) * 384 + j] = (_Float16)v;
    }
  }
}

// ---------------- fc2 (384->96) + residual -> out (f32) ----------------
__global__ __launch_bounds__(256) void fc2_res_kernel(
    const _Float16* __restrict__ h1, const _Float16* __restrict__ w16,
    const float* __restrict__ bias, const float* __restrict__ x1,
    float* __restrict__ out) {
  __shared__ _Float16 sA[64 * 392];     // 784B rows, 16B-aligned
  int rowbase = blockIdx.x * 64;
  {
    const unsigned int sa_base = lds_addr_of(sA);
    const _Float16* gbase = h1 + (size_t)rowbase * 384;
#pragma unroll
    for (int t = 0; t < 12; ++t) {
      int k = threadIdx.x + 256 * t;          // 0..3071
      int r = k / 48, c = k - r * 48;         // 48 x 16B per row
      async_g2l_b128(sa_base + (unsigned)(r * 392 + c * 8) * 2,
                     gbase + r * 384 + c * 8);
    }
  }
  wait_async0();
  __syncthreads();
  int wave = threadIdx.x >> 5, lane = threadIdx.x & 31;
  int g = lane >> 4, l15 = lane & 15;
  for (int t = 0; t < 3; ++t) {
    int tt = wave + 8 * t;              // 24 tiles: 4 M x 6 N
    int mt = tt & 3, nt = tt >> 2;
    int n = nt * 16 + l15;
    v8f c;
    float bv = bias[n];
#pragma unroll
    for (int r = 0; r < 8; ++r) c[r] = bv;
#pragma unroll
    for (int kb = 0; kb < 384; kb += 32) {
      v16h a, bf;
      const _Float16* wr = w16 + (size_t)n * 384 + kb;
      __builtin_prefetch(wr + 384, 0, 0);
#pragma unroll
      for (int i = 0; i < 16; ++i) a[i] = sA[(mt * 16 + l15) * 392 + kb + kof(i, g)];
#pragma unroll
      for (int i = 0; i < 16; ++i) bf[i] = wr[kof(i, g)];
      c = wmma_f16(a, bf, c);
    }
#pragma unroll
    for (int r = 0; r < 8; ++r) {
      int m = mt * 16 + r + (g << 3);
      size_t off = (size_t)(rowbase + m) * CDIM + n;
      out[off] = x1[off] + c[r];                   // residual 2
    }
  }
}

// ---------------- host ----------------
extern "C" void kernel_launch(void* const* d_in, const int* in_sizes, int n_in,
                              void* d_out, int out_size, void* d_ws, size_t ws_size,
                              hipStream_t stream) {
  const float* x      = (const float*)d_in[0];
  const float* maskm  = (const float*)d_in[1];
  const int*   relidx = (const int*)d_in[2];
  const float* qkv_w  = (const float*)d_in[3];
  const float* qkv_b  = (const float*)d_in[4];
  const float* proj_w = (const float*)d_in[5];
  const float* proj_b = (const float*)d_in[6];
  const float* n1g    = (const float*)d_in[7];
  const float* n1b    = (const float*)d_in[8];
  const float* n2g    = (const float*)d_in[9];
  const float* n2b    = (const float*)d_in[10];
  const float* fc1_w  = (const float*)d_in[11];
  const float* fc1_b  = (const float*)d_in[12];
  const float* fc2_w  = (const float*)d_in[13];
  const float* fc2_b  = (const float*)d_in[14];
  const float* rpb    = (const float*)d_in[15];
  float* out = (float*)d_out;

  // workspace carve-up (256B aligned)
  char* ws = (char*)d_ws;
  size_t off = 0;
  auto carve = [&](size_t bytes) {
    char* p = ws + off;
    off = (off + bytes + 255) & ~(size_t)255;
    return p;
  };
  _Float16* qkv_w16  = (_Float16*)carve((size_t)288 * 96 * 2);
  _Float16* proj_w16 = (_Float16*)carve((size_t)96 * 96 * 2);
  _Float16* fc1_w16  = (_Float16*)carve((size_t)384 * 96 * 2);
  _Float16* fc2_w16  = (_Float16*)carve((size_t)96 * 384 * 2);
  _Float16* xw       = (_Float16*)carve((size_t)NWIN * NTOK * CDIM * 2);
  float*    x1       = (float*)   carve((size_t)LTOK * CDIM * 4);
  _Float16* ln2b     = (_Float16*)carve((size_t)LTOK * CDIM * 2);
  _Float16* h1       = (_Float16*)carve((size_t)LTOK * 384 * 2);
  (void)ws_size; (void)n_in; (void)in_sizes; (void)out_size;

  // weights -> f16
  cvt_f16_kernel<<<(288 * 96 + 255) / 256, 256, 0, stream>>>(qkv_w, qkv_w16, 288 * 96);
  cvt_f16_kernel<<<(96 * 96 + 255) / 256, 256, 0, stream>>>(proj_w, proj_w16, 96 * 96);
  cvt_f16_kernel<<<(384 * 96 + 255) / 256, 256, 0, stream>>>(fc1_w, fc1_w16, 384 * 96);
  cvt_f16_kernel<<<(96 * 384 + 255) / 256, 256, 0, stream>>>(fc2_w, fc2_w16, 96 * 384);

  // LN1 + shift + partition
  ln1_shift_part_kernel<<<LTOK / 8, 256, 0, stream>>>(x, n1g, n1b, xw);

  // fused windowed attention (+ proj + residual-1)
  swin_attn_kernel<<<NWIN, 256, 0, stream>>>(xw, qkv_w16, qkv_b, proj_w16, proj_b,
                                             rpb, relidx, maskm, x, x1);

  // LN2
  ln2_kernel<<<LTOK / 8, 256, 0, stream>>>(x1, n2g, n2b, ln2b);

  // MLP
  fc1_gelu_kernel<<<LTOK / 64, 256, 0, stream>>>(ln2b, fc1_w16, fc1_b, h1);
  fc2_res_kernel<<<LTOK / 64, 256, 0, stream>>>(h1, fc2_w16, fc2_b, x1, out);
}
